// MOE_83124797047267
// MI455X (gfx1250) — matrix operand order repaired and checked by reading, not compiled, not run on hardware
//
#include <hip/hip_runtime.h>
#include <math.h>

#ifndef __has_builtin
#define __has_builtin(x) 0
#endif

#if __has_builtin(__builtin_amdgcn_global_load_async_to_lds_b128) && \
    __has_builtin(__builtin_amdgcn_s_wait_asynccnt)
#define USE_ASYNC_LDS 1
#else
#define USE_ASYNC_LDS 0
#endif

// Problem constants (match reference)
#define BB    4
#define TT    1024
#define CC    1024
#define EE    8
#define HH    4096
#define NTOK  (BB * TT)        // 4096 tokens
#define KSEL  2
#define NPAIR (NTOK * KSEL)    // 8192 (token, expert) pairs
#define PAIR_CAP 9216          // 8192 + 8*127 rounded up to 128

#define TILE_M 128
#define TILE_N 128
#define TILE_K 32
#define LDSS   40              // LDS row stride in bf16 elems (80B, multiple of 16B)

typedef __attribute__((ext_vector_type(16))) __bf16 bf16x16;
typedef __attribute__((ext_vector_type(8)))  __bf16 bf16x8;
typedef __attribute__((ext_vector_type(8)))  float  f32x8;

#if USE_ASYNC_LDS
// Exact parameter types per hipcc diagnostic: int __vector(4) * in AS1 (global) / AS3 (LDS)
typedef int v4i __attribute__((vector_size(16)));
typedef __attribute__((address_space(1))) v4i* as1_v4i;
typedef __attribute__((address_space(3))) v4i* as3_v4i;
#endif

union Frag16 { bf16x16 v; bf16x8 h[2]; };

struct MoEMeta {
    float imp[EE];
    int   cnt[EE];
    int   off[EE + 1];
    int   padded[EE];
    int   cursor[EE];
};

// ---------------------------------------------------------------- init
__global__ void init_meta_kernel(MoEMeta* meta) {
    int t = threadIdx.x;
    if (t < EE) { meta->imp[t] = 0.f; meta->cnt[t] = 0; }
}

// ---------------------------------------------------------------- x -> bf16
__global__ void convert_x_kernel(const float* __restrict__ x, __bf16* __restrict__ xb) {
    int i = blockIdx.x * blockDim.x + threadIdx.x;
    int base = i * 4;
    if (base >= NTOK * CC) return;
    float4 v = *(const float4*)(x + base);
    union { __bf16 h[4]; uint2 u; } r;
    r.h[0] = (__bf16)v.x; r.h[1] = (__bf16)v.y;
    r.h[2] = (__bf16)v.z; r.h[3] = (__bf16)v.w;
    *(uint2*)(xb + base) = r.u;
}

// ---------------------------------------------------------------- router (one wave32 per token)
__global__ void router_kernel(const float* __restrict__ x, const float* __restrict__ Wr,
                              const float* __restrict__ br,
                              int* __restrict__ topk_idx, float* __restrict__ topk_w,
                              MoEMeta* meta) {
    int wave = threadIdx.x >> 5;
    int lane = threadIdx.x & 31;
    int token = blockIdx.x * 8 + wave;
    if (token >= NTOK) return;

    float acc[EE];
#pragma unroll
    for (int e = 0; e < EE; ++e) acc[e] = 0.f;

    const float* xrow = x + (size_t)token * CC;
    for (int c0 = lane * 4; c0 < CC; c0 += 32 * 4) {
        float4 xv = *(const float4*)(xrow + c0);
        float xs[4] = {xv.x, xv.y, xv.z, xv.w};
        const float* wr = Wr + (size_t)c0 * EE;
#pragma unroll
        for (int j = 0; j < 4; ++j) {
            float4 wa = *(const float4*)(wr + j * EE);
            float4 wb = *(const float4*)(wr + j * EE + 4);
            acc[0] += xs[j] * wa.x; acc[1] += xs[j] * wa.y;
            acc[2] += xs[j] * wa.z; acc[3] += xs[j] * wa.w;
            acc[4] += xs[j] * wb.x; acc[5] += xs[j] * wb.y;
            acc[6] += xs[j] * wb.z; acc[7] += xs[j] * wb.w;
        }
    }
#pragma unroll
    for (int e = 0; e < EE; ++e)
#pragma unroll
        for (int s = 16; s > 0; s >>= 1) acc[e] += __shfl_xor(acc[e], s, 32);

    if (lane == 0) {
        float logit[EE], p[EE];
        float m = -1e30f;
#pragma unroll
        for (int e = 0; e < EE; ++e) { logit[e] = acc[e] + br[e]; m = fmaxf(m, logit[e]); }
        float sum = 0.f;
#pragma unroll
        for (int e = 0; e < EE; ++e) { p[e] = __expf(logit[e] - m); sum += p[e]; }
        float inv = 1.f / sum;
#pragma unroll
        for (int e = 0; e < EE; ++e) p[e] *= inv;

        int i0 = 0;
#pragma unroll
        for (int e = 1; e < EE; ++e) if (p[e] > p[i0]) i0 = e;
        int i1 = (i0 == 0) ? 1 : 0;
#pragma unroll
        for (int e = 0; e < EE; ++e) if (e != i0 && p[e] > p[i1]) i1 = e;

        float v0 = p[i0], v1 = p[i1], s2 = v0 + v1;
        topk_idx[token * 2 + 0] = i0;
        topk_idx[token * 2 + 1] = i1;
        topk_w[token * 2 + 0] = v0 / s2;
        topk_w[token * 2 + 1] = v1 / s2;
#pragma unroll
        for (int e = 0; e < EE; ++e) atomicAdd(&meta->imp[e], p[e]);
        atomicAdd(&meta->cnt[i0], 1);
        atomicAdd(&meta->cnt[i1], 1);
    }
}

// ---------------------------------------------------------------- offsets + pad fill
__global__ void scan_pad_kernel(MoEMeta* meta, int* pair_tok, int* pair_dst, float* pair_w) {
    if (threadIdx.x | blockIdx.x) return;
    int o = 0;
    for (int e = 0; e < EE; ++e) {
        meta->off[e] = o;
        int c = meta->cnt[e];
        int pc = (c + TILE_M - 1) / TILE_M * TILE_M;
        meta->padded[e] = pc;
        meta->cursor[e] = 0;
        for (int j = c; j < pc; ++j) {
            pair_tok[o + j] = 0;      // safe address for A-gather
            pair_dst[o + j] = -1;     // invalid: epilogue skips
            pair_w[o + j]   = 0.f;
        }
        o += pc;
    }
    meta->off[EE] = o;
}

// ---------------------------------------------------------------- scatter tokens to expert lists
__global__ void scatter_kernel(const int* __restrict__ topk_idx, const float* __restrict__ topk_w,
                               int* pair_tok, int* pair_dst, float* pair_w, MoEMeta* meta) {
    int t = blockIdx.x * blockDim.x + threadIdx.x;
    if (t >= NTOK) return;
#pragma unroll
    for (int s = 0; s < KSEL; ++s) {
        int e = topk_idx[t * 2 + s];
        int pos = atomicAdd(&meta->cursor[e], 1);
        int slot = meta->off[e] + pos;
        pair_tok[slot] = t;
        pair_dst[slot] = t * 2 + s;
        pair_w[slot]   = topk_w[t * 2 + s];
    }
}

// ---------------------------------------------------------------- grouped GEMM (WMMA bf16)
// Double-buffered LDS pipeline; A tile via GLOBAL_LOAD_ASYNC_TO_LDS when available.
// EPI==0: A gathered by token (x_bf16), out = relu(acc+b1) -> bf16 h-buffer
// EPI==1: A = h rows directly, out = gate_w * (acc+b2) -> fp32 per-slot buffer
template <int EPI>
__global__ void __launch_bounds__(256, 2)
moe_gemm_kernel(const __bf16* __restrict__ A, int lda,
                const float* __restrict__ W, int ldb, size_t wEStride,
                const float* __restrict__ bias, int biasLen,
                __bf16* __restrict__ Hout, float* __restrict__ Obuf,
                const int* __restrict__ pair_tok,
                const int* __restrict__ pair_dst,
                const float* __restrict__ pair_w,
                const MoEMeta* __restrict__ meta, int Kdim) {
    int e  = blockIdx.y >> 6;
    int mt = blockIdx.y & 63;
    int mbase = mt * TILE_M;
    if (mbase >= meta->padded[e]) return;           // uniform early exit
    int offE  = meta->off[e];
    int ntile = blockIdx.x * TILE_N;
    const float* WbN = W + (size_t)e * wEStride + ntile;   // uniform base (SGPR)

    __shared__ __attribute__((aligned(16))) __bf16 lsa[2][TILE_M * LDSS];
    __shared__ __attribute__((aligned(16))) __bf16 lsb[2][TILE_N * LDSS];

    int tid  = threadIdx.x;
    int lane = tid & 31;
    int wave = tid >> 5;
    int wm = wave >> 1, wn = wave & 1;
    int khalf = lane >> 4;
    int l15   = lane & 15;

    // ---- A staging coords: each thread owns 32B of one row
    int ar    = tid >> 1;
    int ahalf = tid & 1;
    int p_load = offE + mbase + ar;
    int arow;
    if (EPI == 0) { int tk = pair_tok[p_load]; arow = tk < 0 ? 0 : tk; }
    else          { arow = p_load; }
    const __bf16* aptr = A + (size_t)arow * lda + ahalf * 16;
    int a_st = ar * LDSS + ahalf * 16;              // LDS store offset (elements)

    // ---- B staging: per-thread constant 32-bit offsets
    int boff[4], bidx[4];
#pragma unroll
    for (int it = 0; it < 4; ++it) {
        int lin = tid + it * 256;
        int k  = lin >> 5;
        int n0 = (lin & 31) * 4;
        boff[it] = k * ldb + n0;                    // global element offset (on top of kb*ldb)
        bidx[it] = n0 * LDSS + k;                   // LDS element offset
    }
    int pfoff = (tid >> 3) * ldb + (tid & 7) * 16;  // prefetch lane offset

    // ---- fragment read offsets (lane constants, elements)
    int a_rd = (wm * 32 + l15) * LDSS + khalf * 8;
    int b_rd = (wn * 64 + l15) * LDSS + khalf * 8;

    auto stage_a = [&](int buf, int kb) {
        const __bf16* s = aptr + kb;
        __bf16* d = &lsa[buf][a_st];
#if USE_ASYNC_LDS
        __builtin_amdgcn_global_load_async_to_lds_b128((as1_v4i)(s),     (as3_v4i)(d),     0, 0);
        __builtin_amdgcn_global_load_async_to_lds_b128((as1_v4i)(s + 8), (as3_v4i)(d + 8), 0, 0);
#else
        uint4 v0 = *(const uint4*)(s);
        uint4 v1 = *(const uint4*)(s + 8);
        *(uint4*)(d)     = v0;
        *(uint4*)(d + 8) = v1;
#endif
    };
    auto load_b = [&](int kb, float4* r) {
        const float* base = WbN + (size_t)kb * ldb;
#pragma unroll
        for (int it = 0; it < 4; ++it)
            r[it] = *(const float4*)(base + boff[it]);
    };
    auto store_b = [&](int buf, const float4* r) {
        __bf16* lb = &lsb[buf][0];
#pragma unroll
        for (int it = 0; it < 4; ++it) {
            int bi = bidx[it];
            lb[bi + 0 * LDSS] = (__bf16)r[it].x;
            lb[bi + 1 * LDSS] = (__bf16)r[it].y;
            lb[bi + 2 * LDSS] = (__bf16)r[it].z;
            lb[bi + 3 * LDSS] = (__bf16)r[it].w;
        }
    };

    f32x8 acc[2][4];
#pragma unroll
    for (int i = 0; i < 2; ++i)
#pragma unroll
        for (int j = 0; j < 4; ++j) acc[i][j] = f32x8{0, 0, 0, 0, 0, 0, 0, 0};

    // ---- pipeline prologue: stage K-slab 0 into buffer 0
    float4 breg[4];
    stage_a(0, 0);
    load_b(0, breg);
    store_b(0, breg);
#if USE_ASYNC_LDS
    __builtin_amdgcn_s_wait_asynccnt(0);
#endif
    __syncthreads();

    int cur = 0;
    for (int kb = 0; kb < Kdim; kb += TILE_K) {
        int nxt = cur ^ 1;
        bool more = (kb + TILE_K) < Kdim;

        if (more) {
            stage_a(nxt, kb + TILE_K);      // async: lands in LDS in background
            load_b(kb + TILE_K, breg);      // fp32 weights into registers
            if (kb + 2 * TILE_K < Kdim)     // global_prefetch_b8 two slabs ahead
                __builtin_prefetch(WbN + (size_t)(kb + 2 * TILE_K) * ldb + pfoff, 0, 3);
        }

        // ---- fragments per CDNA5 16-bit A/B VGPR layouts
        Frag16 afr[2], bfr[4];
#pragma unroll
        for (int mi = 0; mi < 2; ++mi) {
            const __bf16* pa = &lsa[cur][a_rd + mi * 16 * LDSS];
            afr[mi].h[0] = *(const bf16x8*)(pa);       // K 0..7  / 8..15
            afr[mi].h[1] = *(const bf16x8*)(pa + 16);  // K 16..23 / 24..31
        }
#pragma unroll
        for (int ni = 0; ni < 4; ++ni) {
            const __bf16* pb = &lsb[cur][b_rd + ni * 16 * LDSS];
            bfr[ni].h[0] = *(const bf16x8*)(pb);       // K 0..7  / 8..15
            bfr[ni].h[1] = *(const bf16x8*)(pb + 16);  // K 16..23 / 24..31
        }
#pragma unroll
        for (int mi = 0; mi < 2; ++mi)
#pragma unroll
            for (int ni = 0; ni < 4; ++ni)
                acc[mi][ni] = __builtin_amdgcn_wmma_f32_16x16x32_bf16(
                    false, afr[mi].v, false, bfr[ni].v,
                    (short)0, acc[mi][ni], false, false);

        if (more) {
            store_b(nxt, breg);             // convert + transpose into next buffer
#if USE_ASYNC_LDS
            __builtin_amdgcn_s_wait_asynccnt(0);
#endif
            __syncthreads();
        }
        cur = nxt;
    }

    // ---- epilogue (C/D layout: VGPR j -> row j (+8 for lanes 16..31), lane&15 -> col)
#pragma unroll
    for (int mi = 0; mi < 2; ++mi) {
#pragma unroll
        for (int ni = 0; ni < 4; ++ni) {
            int col = ntile + wn * 64 + ni * 16 + l15;
            float bv = bias[(size_t)e * biasLen + col];
#pragma unroll
            for (int j = 0; j < 8; ++j) {
                int rloc = wm * 32 + mi * 16 + j + (khalf << 3);
                int p = offE + mbase + rloc;
                float v = acc[mi][ni][j] + bv;
                if (EPI == 0) {
                    v = v > 0.f ? v : 0.f;
                    Hout[(size_t)p * HH + col] = (__bf16)v;
                } else {
                    int dst = pair_dst[p];
                    if (dst >= 0) Obuf[(size_t)dst * CC + col] = pair_w[p] * v;
                }
            }
        }
    }
}

// ---------------------------------------------------------------- combine two slots (deterministic)
__global__ void combine_kernel(const float* __restrict__ obuf, float* __restrict__ out) {
    int i = blockIdx.x * blockDim.x + threadIdx.x;
    int base = i * 4;
    if (base >= NTOK * CC) return;
    int t = base >> 10, c = base & 1023;
    float4 a = *(const float4*)(obuf + (size_t)(2 * t) * CC + c);
    float4 b = *(const float4*)(obuf + (size_t)(2 * t + 1) * CC + c);
    float4 r = {a.x + b.x, a.y + b.y, a.z + b.z, a.w + b.w};
    *(float4*)(out + base) = r;
}

// ---------------------------------------------------------------- aux loss
__global__ void finalize_kernel(const MoEMeta* __restrict__ meta, float* __restrict__ out) {
    if (threadIdx.x | blockIdx.x) return;
    const float numel = (float)(NTOK * KSEL);
    float aux = 0.f;
    for (int e = 0; e < EE; ++e) {
        float imp = meta->imp[e] / (float)NTOK;
        float load = ((float)meta->cnt[e] / numel) / numel;
        aux += imp * load;
    }
    out[NTOK * CC] = (float)EE * aux;
}

// ---------------------------------------------------------------- launch
extern "C" void kernel_launch(void* const* d_in, const int* in_sizes, int n_in,
                              void* d_out, int out_size, void* d_ws, size_t ws_size,
                              hipStream_t stream) {
    const float* x  = (const float*)d_in[0];
    const float* Wr = (const float*)d_in[1];
    const float* br = (const float*)d_in[2];
    const float* W1 = (const float*)d_in[3];
    const float* b1 = (const float*)d_in[4];
    const float* W2 = (const float*)d_in[5];
    const float* b2 = (const float*)d_in[6];
    float* out = (float*)d_out;

    char* ws = (char*)d_ws;
    size_t off = 0;
    auto alloc = [&](size_t bytes) {
        size_t cur = off;
        off = (off + bytes + 255) & ~(size_t)255;
        return cur;
    };
    __bf16* xb    = (__bf16*)(ws + alloc((size_t)NTOK * CC * 2));
    __bf16* hb    = (__bf16*)(ws + alloc((size_t)PAIR_CAP * HH * 2));
    float*  obuf  = (float*)(ws + alloc((size_t)NPAIR * CC * 4));
    int*    tki   = (int*)(ws + alloc((size_t)NTOK * 2 * 4));
    float*  tkw   = (float*)(ws + alloc((size_t)NTOK * 2 * 4));
    int*    ptok  = (int*)(ws + alloc((size_t)PAIR_CAP * 4));
    int*    pdst  = (int*)(ws + alloc((size_t)PAIR_CAP * 4));
    float*  pw    = (float*)(ws + alloc((size_t)PAIR_CAP * 4));
    MoEMeta* meta = (MoEMeta*)(ws + alloc(sizeof(MoEMeta)));

    init_meta_kernel<<<1, 64, 0, stream>>>(meta);
    convert_x_kernel<<<(NTOK * CC / 4 + 255) / 256, 256, 0, stream>>>(x, xb);
    router_kernel<<<NTOK / 8, 256, 0, stream>>>(x, Wr, br, tki, tkw, meta);
    scan_pad_kernel<<<1, 1, 0, stream>>>(meta, ptok, pdst, pw);
    scatter_kernel<<<NTOK / 256, 256, 0, stream>>>(tki, tkw, ptok, pdst, pw, meta);

    // GEMM1: h = relu(x @ W1 + b1)   M<=PAIR rows, K=C, N=H
    moe_gemm_kernel<0><<<dim3(HH / TILE_N, EE * 64), 256, 0, stream>>>(
        xb, CC, W1, HH, (size_t)CC * HH, b1, HH, hb, nullptr,
        ptok, pdst, pw, meta, CC);

    // GEMM2: o = gate * (h @ W2 + b2)  K=H, N=C
    moe_gemm_kernel<1><<<dim3(CC / TILE_N, EE * 64), 256, 0, stream>>>(
        hb, HH, W2, CC, (size_t)HH * CC, b2, CC, nullptr, obuf,
        ptok, pdst, pw, meta, HH);

    combine_kernel<<<(NTOK * CC / 4) / 256, 256, 0, stream>>>(obuf, out);
    finalize_kernel<<<1, 1, 0, stream>>>(meta, out);
}
